// _DeepTransition_69148973465952
// MI455X (gfx1250) — compile-verified, matrix operand order checked by reading, not run on hardware
//
#include <hip/hip_runtime.h>
#include <hip/hip_bf16.h>

// ---------------- problem constants (match reference) ----------------
#define N_NODES 20000
#define C_IN    128
#define C_OUT   64
#define T_STEPS 8
#define N_EDGES 160000
#define THRESH  0.5f
#define LENS    0.5f
#define DECAY   0.2f
#define EPS_BN  1e-5f

#define M_ROWS  (N_NODES * T_STEPS)        // 160000 GEMM rows, = node*8 + t
#define ROW_K   C_IN                       // 128

// ---------------- WMMA vector types (gfx1250, wave32) ----------------
typedef __attribute__((ext_vector_type(16))) __bf16          v16bf;
typedef __attribute__((ext_vector_type(8)))  float           v8f;
typedef __attribute__((ext_vector_type(8)))  unsigned short  ush8;

union FragU {            // 32 bytes: 16 bf16 lanes-worth of A/B fragment
    ush8  h[2];
    v16bf v;
};

// fp32 -> bf16 via native cast (lowers to HW cvt on gfx1250; RNE semantics)
__device__ __forceinline__ unsigned short f2bf(float f) {
    __bf16 b = (__bf16)f;
    return __builtin_bit_cast(unsigned short, b);
}

// 8 contiguous f32 -> bf16 into fragment elements [base, base+8)
__device__ __forceinline__ void cvt8_into(const float* p, v16bf& v, int base) {
    float4 a = ((const float4*)p)[0];
    float4 b = ((const float4*)p)[1];
    v[base + 0] = (__bf16)a.x; v[base + 1] = (__bf16)a.y;
    v[base + 2] = (__bf16)a.z; v[base + 3] = (__bf16)a.w;
    v[base + 4] = (__bf16)b.x; v[base + 5] = (__bf16)b.y;
    v[base + 6] = (__bf16)b.z; v[base + 7] = (__bf16)b.w;
}

// ---------------- 1. per-channel stats: mean, rsqrt(var+eps) ----------------
__global__ void stats_kernel(const float* __restrict__ x,
                             float* __restrict__ mean, float* __restrict__ scale) {
    const int c   = blockIdx.x;          // 0..127
    const int tid = threadIdx.x;         // 0..255
    float s = 0.f, ss = 0.f;
    for (int n = tid; n < N_NODES; n += 256) {
        const float* p = x + (size_t)n * (C_IN * T_STEPS) + (size_t)c * T_STEPS;
        float4 a = ((const float4*)p)[0];
        float4 b = ((const float4*)p)[1];
        s  += a.x + a.y + a.z + a.w + b.x + b.y + b.z + b.w;
        ss += a.x*a.x + a.y*a.y + a.z*a.z + a.w*a.w
            + b.x*b.x + b.y*b.y + b.z*b.z + b.w*b.w;
    }
    __shared__ float sh_s[256], sh_q[256];
    sh_s[tid] = s; sh_q[tid] = ss;
    __syncthreads();
    for (int o = 128; o > 0; o >>= 1) {
        if (tid < o) { sh_s[tid] += sh_s[tid + o]; sh_q[tid] += sh_q[tid + o]; }
        __syncthreads();
    }
    if (tid == 0) {
        const float inv = 1.0f / (float)(N_NODES * T_STEPS);
        float m = sh_s[0] * inv;
        float v = sh_q[0] * inv - m * m;
        mean[c]  = m;
        scale[c] = rsqrtf(v + EPS_BN);
    }
}

// ---------------- 2. fused BN + LIF -> bf16 spikes, layout [node*8+t, c] ----------------
__global__ void bnorm_lif_kernel(const float* __restrict__ x,
                                 const float* __restrict__ gamma,
                                 const float* __restrict__ beta,
                                 const float* __restrict__ mean,
                                 const float* __restrict__ scale,
                                 unsigned short* __restrict__ s_out) {
    int idx = blockIdx.x * blockDim.x + threadIdx.x;     // node*128 + c
    if (idx >= N_NODES * C_IN) return;
    const int node = idx >> 7;
    const int c    = idx & 127;
    const float* p = x + (size_t)node * (C_IN * T_STEPS) + (size_t)c * T_STEPS;
    float4 a = ((const float4*)p)[0];
    float4 b = ((const float4*)p)[1];
    float h[T_STEPS] = {a.x, a.y, a.z, a.w, b.x, b.y, b.z, b.w};
    const float g = gamma[c], be = beta[c], m = mean[c], sc = scale[c];
    float mem = 0.f, sp = 0.f;
    unsigned short* srow = s_out + (size_t)node * (T_STEPS * C_IN) + c;
#pragma unroll
    for (int t = 0; t < T_STEPS; ++t) {
        float hv = g * (h[t] - m) * sc + be;
        mem = mem * DECAY * (1.f - sp) + hv;
        sp  = (mem > THRESH) ? 1.f : 0.f;
        srow[(size_t)t * C_IN] = (sp > 0.5f) ? (unsigned short)0x3F80 : (unsigned short)0;
    }
}

// ---------------- 3. W0/W1 fp32 -> bf16 ----------------
__global__ void convw_kernel(const float* __restrict__ W0, const float* __restrict__ W1,
                             unsigned short* __restrict__ w0b, unsigned short* __restrict__ w1b) {
    int i = blockIdx.x * blockDim.x + threadIdx.x;
    if (i < C_OUT * C_IN) { w0b[i] = f2bf(W0[i]); w1b[i] = f2bf(W1[i]); }
}

// ---------------- 4. zero fp32 region (grid-stride) ----------------
__global__ void zero_kernel(float* __restrict__ p, int n) {
    for (int i = blockIdx.x * blockDim.x + threadIdx.x; i < n; i += gridDim.x * blockDim.x)
        p[i] = 0.f;
}

// ---------------- 5. degree accumulation ----------------
__global__ void deg_kernel(const long long* __restrict__ ei, float* __restrict__ deg) {
    int e = blockIdx.x * blockDim.x + threadIdx.x;
    if (e < N_EDGES) {
        long long dst = ei[N_EDGES + e];
        atomicAdd(&deg[dst], 1.0f);
    }
}

// ---------------- 6. dinv = deg>0 ? rsqrt(max(deg,1e-12)) : 0 ----------------
__global__ void dinv_kernel(const float* __restrict__ deg, float* __restrict__ dinv) {
    int n = blockIdx.x * blockDim.x + threadIdx.x;
    if (n < N_NODES) {
        float d = deg[n];
        dinv[n] = (d > 0.f) ? rsqrtf(fmaxf(d, 1e-12f)) : 0.f;
    }
}

// ---------------- 7. edge scatter: agg[dst] += w * s[src] ----------------
// one block per edge, 256 threads x 4 elems = 1024 (= T*C) elems per row.
// spikes are exactly 0/1 in bf16 (0x0000 / 0x3F80): skip zeros, add w for ones.
// the 40 MB spike tensor stays L2-resident (192 MB L2), so the gather is cheap;
// the f32 atomic-add stream to agg is the true bottleneck of the whole op.
__global__ void scatter_kernel(const long long* __restrict__ ei,
                               const float* __restrict__ dinv,
                               const unsigned short* __restrict__ s,
                               float* __restrict__ agg) {
    const int e   = blockIdx.x;
    const int tid = threadIdx.x;
    const long long src = ei[e];
    const long long dst = ei[N_EDGES + e];
    const float w = dinv[src] * dinv[dst];
    if (w == 0.f) return;                    // block-uniform branch
    const unsigned short* srow = s + (size_t)src * (T_STEPS * C_IN) + tid * 4;
    float* arow = agg + (size_t)dst * (T_STEPS * C_IN) + tid * 4;
    uint2 pv = *(const uint2*)srow;          // 4 bf16 spikes
    if (pv.x & 0x0000FFFFu) atomicAdd(&arow[0], w);
    if (pv.x & 0xFFFF0000u) atomicAdd(&arow[1], w);
    if (pv.y & 0x0000FFFFu) atomicAdd(&arow[2], w);
    if (pv.y & 0xFFFF0000u) atomicAdd(&arow[3], w);
}

// ---------------- 8. WMMA GEMM: out = S@W0^T + AGG@W1^T + b ----------------
// one wave32 per 16x16 output tile: mtile in [0,10000), ctile in [0,4).
// K=128 -> 4 k-steps of v_wmma_f32_16x16x32_bf16 per input matrix (8 WMMAs/wave).
// ISA 16-bit A/B layout: lanes 0-15 take K-half 0, lanes 16-31 take K-half +8;
// fragment elems 0..7 = K+{0..7}, elems 8..15 = K+{16..23} of each 32-wide step.
__global__ void __launch_bounds__(256) gemm_kernel(const unsigned short* __restrict__ s,
                                                   const float* __restrict__ agg,
                                                   const unsigned short* __restrict__ w0b,
                                                   const unsigned short* __restrict__ w1b,
                                                   const float* __restrict__ bias,
                                                   float* __restrict__ out) {
    const int wave  = (blockIdx.x * blockDim.x + threadIdx.x) >> 5;
    const int lane  = threadIdx.x & 31;
    const int mtile = wave >> 2;            // 0..9999
    const int ctile = wave & 3;             // 0..3
    const int half  = lane >> 4;            // K-half select per ISA A/B layout
    const int lrow  = lane & 15;

    const int m = mtile * 16 + lrow;        // A row  (= node*8 + t)
    const int d = ctile * 16 + lrow;        // B column / output channel

    const unsigned short* srow = s   + (size_t)m * ROW_K;
    const float*          arow = agg + (size_t)m * ROW_K;
    const unsigned short* w0r  = w0b + (size_t)d * ROW_K;  // W[d, :] row = B column d
    const unsigned short* w1r  = w1b + (size_t)d * ROW_K;

    v8f acc = {};
#pragma unroll
    for (int k = 0; k < 4; ++k) {           // spikes @ W0^T
        const int cA = k * 32 + half * 8;
        FragU fa, fb;
        fa.h[0] = *(const ush8*)(srow + cA);
        fa.h[1] = *(const ush8*)(srow + cA + 16);
        fb.h[0] = *(const ush8*)(w0r  + cA);
        fb.h[1] = *(const ush8*)(w0r  + cA + 16);
        acc = __builtin_amdgcn_wmma_f32_16x16x32_bf16(false, fa.v, false, fb.v,
                                                      (short)0, acc, false, false);
    }
#pragma unroll
    for (int k = 0; k < 4; ++k) {           // agg @ W1^T (HW f32->bf16 cvt in-regs)
        const int cA = k * 32 + half * 8;
        FragU fa, fb;
        cvt8_into(arow + cA,      fa.v, 0);
        cvt8_into(arow + cA + 16, fa.v, 8);
        fb.h[0] = *(const ush8*)(w1r  + cA);
        fb.h[1] = *(const ush8*)(w1r  + cA + 16);
        acc = __builtin_amdgcn_wmma_f32_16x16x32_bf16(false, fa.v, false, fb.v,
                                                      (short)0, acc, false, false);
    }

    const float bv = bias[d];
    // D layout: lanes 0-15 -> M = r, lanes 16-31 -> M = r+8; N = lane%16
#pragma unroll
    for (int r = 0; r < 8; ++r) {
        const int mel  = mtile * 16 + half * 8 + r;  // global row = node*8 + t
        const int node = mel >> 3;
        const int t    = mel & 7;
        out[(size_t)node * (C_OUT * T_STEPS) + d * T_STEPS + t] = acc[r] + bv;
    }
}

// ---------------- host-side launch ----------------
extern "C" void kernel_launch(void* const* d_in, const int* in_sizes, int n_in,
                              void* d_out, int out_size, void* d_ws, size_t ws_size,
                              hipStream_t stream) {
    const float*     x     = (const float*)d_in[0];
    const long long* ei    = (const long long*)d_in[1];   // int64 [2, E]
    const float*     gamma = (const float*)d_in[2];
    const float*     beta  = (const float*)d_in[3];
    const float*     W0    = (const float*)d_in[4];
    const float*     W1    = (const float*)d_in[5];
    const float*     bias  = (const float*)d_in[6];
    float*           out   = (float*)d_out;

    // workspace layout (128B-aligned regions)
    char* ws = (char*)d_ws;
    float*          mean  = (float*)(ws + 0);                        // 128 f32
    float*          scale = (float*)(ws + 512);                      // 128 f32
    unsigned short* w0b   = (unsigned short*)(ws + 1024);            // 8192 bf16
    unsigned short* w1b   = (unsigned short*)(ws + 17408);           // 8192 bf16
    float*          deg   = (float*)(ws + 33792);                    // 20000 f32
    float*          dinv  = (float*)(ws + 113792);                   // 20000 f32
    unsigned short* sbuf  = (unsigned short*)(ws + 193792);          // 160000*128 bf16 (40.96 MB)
    float*          agg   = (float*)(ws + 41153792ull);              // 160000*128 f32  (81.92 MB)

    // 1. channel statistics
    stats_kernel<<<C_IN, 256, 0, stream>>>(x, mean, scale);
    // 2. BN + LIF -> bf16 spikes
    bnorm_lif_kernel<<<(N_NODES * C_IN + 255) / 256, 256, 0, stream>>>(
        x, gamma, beta, mean, scale, sbuf);
    // 3. weights to bf16
    convw_kernel<<<(C_OUT * C_IN + 255) / 256, 256, 0, stream>>>(W0, W1, w0b, w1b);
    // 4. zero degree + aggregation buffers
    zero_kernel<<<79, 256, 0, stream>>>(deg, N_NODES);
    zero_kernel<<<4096, 256, 0, stream>>>(agg, M_ROWS * ROW_K);
    // 5. degrees
    deg_kernel<<<(N_EDGES + 255) / 256, 256, 0, stream>>>(ei, deg);
    // 6. dinv
    dinv_kernel<<<(N_NODES + 255) / 256, 256, 0, stream>>>(deg, dinv);
    // 7. edge scatter
    scatter_kernel<<<N_EDGES, 256, 0, stream>>>(ei, dinv, sbuf, agg);
    // 8. WMMA GEMM + bias, scatter-store to [n, d, t]
    //    total waves = 10000 mtiles * 4 ctiles = 40000 -> 5000 blocks * 8 waves
    gemm_kernel<<<5000, 256, 0, stream>>>(sbuf, agg, w0b, w1b, bias, out);
}
